// GraphAttentionLayer_54400055771193
// MI455X (gfx1250) — matrix operand order repaired and checked by reading, not compiled, not run on hardware
//
#include <hip/hip_runtime.h>

typedef __attribute__((ext_vector_type(16))) _Float16 v16h;
typedef __attribute__((ext_vector_type(8)))  float    v8f;
typedef __attribute__((ext_vector_type(4)))  float    v4f;

#define GAT_ALPHA 0.2f
#define BIGF      2048.0f
#define L2E       1.4426950408889634f

constexpr int Bb = 8;
constexpr int Nn = 2048;
constexpr int Ff = 128;

// --------------------------------------------------------------------------
// Kernel 1: Wh = (h + PE) @ W  (f16 WMMA, f32 accum)
//   - stages W transposed (f16) in LDS
//   - writes Wh^T as f16 [b][o][n]  (feeds B-fragments of kernel 3)
//   - computes s1[b][n] = Wh . a1, s2[b][n] = Wh . a2 from the accumulators
// --------------------------------------------------------------------------
__global__ __launch_bounds__(256)
void k_wh(const float* __restrict__ h, const float* __restrict__ W,
          const float* __restrict__ a, _Float16* __restrict__ wht,
          float* __restrict__ s1, float* __restrict__ s2)
{
    __shared__ _Float16 Wt[128 * 144];   // transposed W, padded row stride (288B = 9*32B)

    const int tid = threadIdx.x;
    for (int idx = tid; idx < 128 * 128; idx += 256) {
        const int k = idx >> 7, o = idx & 127;
        Wt[o * 144 + k] = (_Float16)W[idx];
    }
    __syncthreads();

    const int wave = tid >> 5, lane = tid & 31;
    const int tile = blockIdx.x * 8 + wave;      // 0..1023 (128 tiles per batch)
    const int b    = tile >> 7;
    const int ibase = (tile & 127) << 4;
    const int row = lane & 15, hi = lane >> 4;
    const int n = ibase + row;

    const float decay = 1.0f + __expf(-(float)n * 0.01f);
    const float* __restrict__ hrow = h + ((size_t)b * Nn + n) * Ff;

    v8f acc[8] = {};

    for (int kt = 0; kt < 4; ++kt) {
        v16h afrag;
        #pragma unroll
        for (int e = 0; e < 16; ++e) {
            const int koff = e + (hi ? 16 : 0);
            const int d = kt * 32 + koff;
            const int d2 = d & ~1;
            const float freq = __expf((float)d2 * -0.07195578415606394f); // -ln(1e4)/128
            const float ang = (float)n * freq;
            const float pe = ((d & 1) ? __cosf(ang) : __sinf(ang)) * decay;
            afrag[e] = (_Float16)(hrow[d] + pe);
        }
        #pragma unroll
        for (int t = 0; t < 8; ++t) {
            const v16h bfrag = *(const v16h*)&Wt[(t * 16 + row) * 144 + kt * 32 + (hi ? 16 : 0)];
            acc[t] = __builtin_amdgcn_wmma_f32_16x16x32_f16(
                false, afrag, false, bfrag, (short)0, acc[t], false, false);
        }
    }

    // Wh^T f16 stores: lane owns column o = t*16 + row, rows ibase + 8*hi + (0..7)
    #pragma unroll
    for (int t = 0; t < 8; ++t) {
        const int o = t * 16 + row;
        __attribute__((aligned(16))) _Float16 tmp[8];
        #pragma unroll
        for (int r = 0; r < 8; ++r) tmp[r] = (_Float16)acc[t][r];
        *(float4*)&wht[((size_t)(b * 128 + o)) * Nn + ibase + 8 * hi] = *(const float4*)tmp;
    }

    // s1/s2: per-row dot with a1/a2 straight from accumulator layout
    float q1[8], q2[8];
    #pragma unroll
    for (int r = 0; r < 8; ++r) { q1[r] = 0.f; q2[r] = 0.f; }
    #pragma unroll
    for (int t = 0; t < 8; ++t) {
        const float a1v = a[t * 16 + row];
        const float a2v = a[128 + t * 16 + row];
        #pragma unroll
        for (int r = 0; r < 8; ++r) { q1[r] += acc[t][r] * a1v; q2[r] += acc[t][r] * a2v; }
    }
    #pragma unroll
    for (int msk = 1; msk <= 8; msk <<= 1) {
        #pragma unroll
        for (int r = 0; r < 8; ++r) {
            q1[r] += __shfl_xor(q1[r], msk, 32);
            q2[r] += __shfl_xor(q2[r], msk, 32);
        }
    }
    if ((lane & 15) == 0) {
        #pragma unroll
        for (int r = 0; r < 8; ++r) {
            s1[(size_t)b * Nn + ibase + 8 * hi + r] = q1[r];
            s2[(size_t)b * Nn + ibase + 8 * hi + r] = q2[r];
        }
    }
}

// --------------------------------------------------------------------------
// Kernel 2 (prep): per batch, s2max = max_j s2; emit log2(e)-scaled array and
// per-row exponent constants with the mask shift folded in:
//   s2c = s2*L2E
//   m̂c_i = leaky(s1c_i + max(s2c))        (upper bound of true row max)
//   c1 = s1c - m̂c - BIG ; c2 = 0.2*s1c - m̂c - BIG
// so that p = exp2( fma(adj, BIG, max(c1 + s2c_j, fma(s2c_j, 0.2, c2))) )
// --------------------------------------------------------------------------
__global__ __launch_bounds__(256)
void k_prep(const float* __restrict__ s1, const float* __restrict__ s2,
            float* __restrict__ c1a, float* __restrict__ c2a,
            float* __restrict__ s2c)
{
    __shared__ float red[256];
    const int b = blockIdx.x, tid = threadIdx.x;
    const float* __restrict__ s2b = s2 + (size_t)b * Nn;

    float mx = -3.4e38f;
    for (int j = tid; j < Nn; j += 256) mx = fmaxf(mx, s2b[j]);
    red[tid] = mx;
    __syncthreads();
    for (int s = 128; s > 0; s >>= 1) {
        if (tid < s) red[tid] = fmaxf(red[tid], red[tid + s]);
        __syncthreads();
    }
    const float s2cmax = red[0] * L2E;

    for (int j = tid; j < Nn; j += 256) {
        const float sc = s2b[j] * L2E;
        s2c[(size_t)b * Nn + j] = sc;
        const float s1c = s1[(size_t)b * Nn + j] * L2E;
        const float x = s1c + s2cmax;
        const float mhat = fmaxf(x, GAT_ALPHA * x);
        c1a[(size_t)b * Nn + j] = s1c - mhat - BIGF;
        c2a[(size_t)b * Nn + j] = GAT_ALPHA * s1c - mhat - BIGF;
    }
}

// --------------------------------------------------------------------------
// Kernel 3: fused masked-softmax attention + attn @ Wh.
// Fixed per-row exponent shift (no online rescaling). 8 waves/block, each
// wave owns a 16-row i-tile; double-buffered LDS Wh^T j-chunk (one barrier
// per iteration). adj streamed NT. Column sums (rows 1..N-1) -> g atomics.
// --------------------------------------------------------------------------
__global__ __launch_bounds__(256)
void k_attn(const float* __restrict__ adj, const _Float16* __restrict__ wht,
            const float* __restrict__ c1a, const float* __restrict__ c2a,
            const float* __restrict__ s2c, float* __restrict__ graw)
{
    __shared__ _Float16 lw[2][128 * 48];  // Wh^T chunk [o][32 j], padded (96B rows)

    const int tid = threadIdx.x, wave = tid >> 5, lane = tid & 31;
    const int b = blockIdx.x >> 4;           // 16 blocks per batch
    const int grp = blockIdx.x & 15;
    const int ibase = grp * 128 + wave * 16;
    const int row = lane & 15, hi = lane >> 4;
    const int irow = ibase + row;
    const int base0 = hi ? 16 : 0;
    const int diagjc = ibase >> 5;           // the one chunk holding j == irow

    const float c1 = c1a[(size_t)b * Nn + irow];
    const float c2 = c2a[(size_t)b * Nn + irow];
    const float* __restrict__ s2cb = s2c + (size_t)b * Nn;

    v8f acc[8] = {};
    float lsum = 0.f;

    for (int jc = 0; jc < Nn / 32; ++jc) {
        const int jbase = jc * 32;
        _Float16* __restrict__ lwb = lw[jc & 1];

        {   // cooperative stage: thread pair per row o (2 x 32B)
            const int o = tid >> 1, seg = tid & 1;
            const float4* src = (const float4*)(wht + ((size_t)(b * 128 + o)) * Nn + jbase + seg * 16);
            float4* dst = (float4*)&lwb[o * 48 + seg * 16];
            dst[0] = src[0]; dst[1] = src[1];
        }
        __syncthreads();   // single barrier: also fences reuse of this buffer
                           // (compute(jc-2) finished before any wave passed sync(jc-1))

        // p = exp2( fma(adj, BIG, max(c1 + s2c_j, fma(s2c_j, 0.2, c2))) )
        const float* __restrict__ adjrow = adj + ((size_t)b * Nn + irow) * Nn + jbase;
        float pv[16];
        #pragma unroll
        for (int g = 0; g < 2; ++g) {
            const int off = base0 + g * 8;
            const v4f av0 = __builtin_nontemporal_load((const v4f*)(adjrow + off));
            const v4f av1 = __builtin_nontemporal_load((const v4f*)(adjrow + off + 4));
            const v4f u0 = *(const v4f*)(s2cb + jbase + off);
            const v4f u1 = *(const v4f*)(s2cb + jbase + off + 4);
            const float ad[8] = {av0[0], av0[1], av0[2], av0[3], av1[0], av1[1], av1[2], av1[3]};
            const float uu[8] = {u0[0], u0[1], u0[2], u0[3], u1[0], u1[1], u1[2], u1[3]};
            #pragma unroll
            for (int u = 0; u < 8; ++u) {
                const float m = fmaxf(c1 + uu[u], fmaf(uu[u], GAT_ALPHA, c2));
                const float arg = fmaf(ad[u], BIGF, m);
                pv[g * 8 + u] = __builtin_amdgcn_exp2f(arg);
            }
        }

        if (jc == diagjc) {      // wave-uniform: force-unmask the diagonal element
            const float sd = s2cb[irow];
            const float mm = fmaxf(c1 + sd, fmaf(sd, GAT_ALPHA, c2));
            const float pnew = __builtin_amdgcn_exp2f(mm + BIGF);
            #pragma unroll
            for (int e = 0; e < 16; ++e) {
                const int off = e + base0;
                if (jbase + off == irow) pv[e] = pnew;
            }
        }

        v16h afrag;
        #pragma unroll
        for (int e = 0; e < 16; ++e) { lsum += pv[e]; afrag[e] = (_Float16)pv[e]; }

        #pragma unroll
        for (int t = 0; t < 8; ++t) {
            const v16h bfrag = *(const v16h*)&lwb[(t * 16 + row) * 48 + (hi ? 16 : 0)];
            acc[t] = __builtin_amdgcn_wmma_f32_16x16x32_f16(
                false, afrag, false, bfrag, (short)0, acc[t], false, false);
        }
    }

    // finalize: h_prime rows = acc / l ; sum rows (excluding global row 0) into g
    lsum += __shfl_xor(lsum, 16, 32);
    const float linv_mine = 1.0f / lsum;
    float li[8];
    #pragma unroll
    for (int r = 0; r < 8; ++r) li[r] = __shfl(linv_mine, r + 8 * hi, 32);

    #pragma unroll
    for (int t = 0; t < 8; ++t) {
        float csum = 0.f;
        #pragma unroll
        for (int r = 0; r < 8; ++r) {
            const float inc = (ibase == 0 && hi == 0 && r == 0) ? 0.f : 1.f; // drop node 0
            csum += acc[t][r] * li[r] * inc;
        }
        csum += __shfl_xor(csum, 16, 32);
        if (hi == 0) atomicAdd(&graw[b * 128 + t * 16 + row], csum);
    }
}

// --------------------------------------------------------------------------
// Kernel 4: g = elu(graw / 2047)   (denom is exactly N-1: diag makes deg>=1)
// --------------------------------------------------------------------------
__global__ void k_final(const float* __restrict__ graw, float* __restrict__ out)
{
    const int i = blockIdx.x * blockDim.x + threadIdx.x;
    if (i < Bb * Ff) {
        const float g = graw[i] * (1.0f / 2047.0f);
        out[i] = (g > 0.f) ? g : (__expf(g) - 1.f);
    }
}

__global__ void k_zero(float* __restrict__ g)
{
    const int i = blockIdx.x * blockDim.x + threadIdx.x;
    if (i < Bb * Ff) g[i] = 0.f;
}

extern "C" void kernel_launch(void* const* d_in, const int* in_sizes, int n_in,
                              void* d_out, int out_size, void* d_ws, size_t ws_size,
                              hipStream_t stream)
{
    const float* h   = (const float*)d_in[0];
    const float* adj = (const float*)d_in[1];
    const float* W   = (const float*)d_in[2];
    const float* a   = (const float*)d_in[3];

    char* ws = (char*)d_ws;
    _Float16* wht = (_Float16*)ws;                                    // 4 MiB: Wh^T f16
    float* s1   = (float*)(ws + 4u * 1024 * 1024);                    // 64 KiB each
    float* s2   = (float*)(ws + 4u * 1024 * 1024 + 1 * 64 * 1024);
    float* c1a  = (float*)(ws + 4u * 1024 * 1024 + 2 * 64 * 1024);
    float* c2a  = (float*)(ws + 4u * 1024 * 1024 + 3 * 64 * 1024);
    float* s2c  = (float*)(ws + 4u * 1024 * 1024 + 4 * 64 * 1024);
    float* graw = (float*)(ws + 4u * 1024 * 1024 + 5 * 64 * 1024);    // 4 KiB

    k_zero <<<4, 256, 0, stream>>>(graw);
    k_wh   <<<128, 256, 0, stream>>>(h, W, a, wht, s1, s2);
    k_prep <<<Bb, 256, 0, stream>>>(s1, s2, c1a, c2a, s2c);
    k_attn <<<Bb * (Nn / 128), 256, 0, stream>>>(adj, wht, c1a, c2a, s2c, graw);
    k_final<<<4, 256, 0, stream>>>(graw, (float*)d_out);
}
